// GEMBConv_43834436223106
// MI455X (gfx1250) — compile-verified, matrix operand order checked by reading, not run on hardware
//
#include <hip/hip_runtime.h>
#include <stdint.h>

typedef __bf16 bf16;
typedef __attribute__((ext_vector_type(16))) bf16 bf16x16;
typedef __attribute__((ext_vector_type(8))) float v8f;

union Frag {
  bf16x16 v;
  uint4 q[2];
};

__device__ __forceinline__ v8f wmma_bf16(bf16x16 a, bf16x16 b, v8f c) {
  return __builtin_amdgcn_wmma_f32_16x16x32_bf16(false, a, false, b, (short)0, c,
                                                 false, false);
}

__device__ __forceinline__ unsigned short f2bf(float x) {
  unsigned u = __builtin_bit_cast(unsigned, x);
  u = u + 0x7FFFu + ((u >> 16) & 1u);   // round-to-nearest-even
  return (unsigned short)(u >> 16);
}
__device__ __forceinline__ unsigned packbf2(float lo, float hi) {
  return (unsigned)f2bf(lo) | ((unsigned)f2bf(hi) << 16);
}

// async DMA: global -> LDS, 16 bytes, tracked by ASYNCcnt
__device__ __forceinline__ void async_b128(unsigned lds_addr, const void* gaddr) {
  asm volatile("global_load_async_to_lds_b128 %0, %1, off"
               :: "v"(lds_addr), "v"(gaddr)
               : "memory");
}

// ---------------- prep kernels ----------------
__global__ void k_cvt(const float* __restrict__ in, unsigned short* __restrict__ out, int n) {
  int i = blockIdx.x * blockDim.x + threadIdx.x;
  if (i < n) out[i] = f2bf(in[i]);
}

// W[K,N] fp32 -> WT[N,K] bf16
__global__ void k_tcvt(const float* __restrict__ W, unsigned short* __restrict__ WT,
                       int K, int N) {
  int i = blockIdx.x * blockDim.x + threadIdx.x;
  if (i < K * N) {
    int n = i / K, k = i % K;
    WT[i] = f2bf(W[(long)k * N + n]);
  }
}

__global__ void k_zero4(float4* __restrict__ p, int n4) {
  int i = blockIdx.x * blockDim.x + threadIdx.x;
  if (i < n4) p[i] = make_float4(0.f, 0.f, 0.f, 0.f);
}

// ---------------- message kernel: gather -> MLP -> scatter-add ----------------
// one wave = 16 edges; block = 4 waves. Weight tiles (16x256 bf16 = 8KB) are
// async-DMA staged into a double-buffered LDS slot shared by all 4 waves.
__global__ __launch_bounds__(128) void k_msg(
    const unsigned short* __restrict__ xb,    // [N,128] bf16
    const int* __restrict__ eidx,             // [2,E] row0=target, row1=source
    const unsigned short* __restrict__ W12T,  // [256+384, 256] bf16 (transposed, contiguous)
    const float* __restrict__ b1,
    const float* __restrict__ b2,
    float* __restrict__ aggr,                 // [N,384] f32
    int E) {
  __shared__ unsigned short h[4][16 * 256];
  __shared__ unsigned short wbuf[2][16 * 256];
  const int tid = threadIdx.x;
  const int wave = tid >> 5;
  const int lane = tid & 31;
  const int m = lane & 15;
  const int hi = lane >> 4;
  const long eBase = ((long)blockIdx.x * 4 + wave) * 16;
  const int trow = eidx[eBase + m];            // target of edge m in tile
  const int srow = eidx[(long)E + eBase + m];  // source of edge m in tile

  // layer-1 A fragments: concat [x_src(128) | x_tgt(128)] -> K=256
  Frag A[8];
#pragma unroll
  for (int kt = 0; kt < 8; ++kt) {
    const int row = (kt < 4) ? srow : trow;
    const int koff = ((kt < 4) ? kt : kt - 4) * 32 + hi * 16;
    const uint4* p = (const uint4*)(xb + (long)row * 128 + koff);
    A[kt].q[0] = p[0];
    A[kt].q[1] = p[1];
  }

  // prefetch weight tile 0 into LDS (each thread moves 64B of the 8KB tile)
  {
    const char* g = (const char*)W12T + (size_t)tid * 64;
    unsigned l = (unsigned)(uintptr_t)&wbuf[0][0] + tid * 64;
#pragma unroll
    for (int q = 0; q < 4; ++q) async_b128(l + q * 16, g + q * 16);
  }

  Frag A2[8];  // layer-2 A fragments (filled at phase switch)

  // 40 weight tiles: 0..15 = layer 1 (256x256), 16..39 = layer 2 (256x384)
  for (int i = 0; i < 40; ++i) {
    const int cur = i & 1;
    if (i + 1 < 40) {
      const char* g = (const char*)W12T + (size_t)(i + 1) * 8192 + (size_t)tid * 64;
      unsigned l = (unsigned)(uintptr_t)&wbuf[(i + 1) & 1][0] + tid * 64;
#pragma unroll
      for (int q = 0; q < 4; ++q) async_b128(l + q * 16, g + q * 16);
      asm volatile("s_wait_asynccnt 4" ::: "memory");  // tile i landed (in-order)
    } else {
      asm volatile("s_wait_asynccnt 0" ::: "memory");
    }
    __syncthreads();  // staged tile visible to all waves

    if (i < 16) {
      // layer 1: [16x256]@[256x256] + b1, ReLU -> h (bf16 in LDS)
      const int ct = i;
      v8f acc = {};
#pragma unroll
      for (int kt = 0; kt < 8; ++kt) {
        Frag B;
        const uint4* p = (const uint4*)&wbuf[cur][m * 256 + kt * 32 + hi * 16];
        B.q[0] = p[0];
        B.q[1] = p[1];
        acc = wmma_bf16(A[kt].v, B.v, acc);
      }
      const float bias = b1[ct * 16 + m];
#pragma unroll
      for (int j = 0; j < 8; ++j) {
        const float v = fmaxf(acc[j] + bias, 0.f);
        h[wave][(j + 8 * hi) * 256 + ct * 16 + m] = f2bf(v);
      }
    } else {
      if (i == 16) {
        // D->A relayout of h (all h writes fenced by the barrier above)
#pragma unroll
        for (int kt = 0; kt < 8; ++kt) {
          const uint4* p = (const uint4*)&h[wave][m * 256 + kt * 32 + hi * 16];
          A2[kt].q[0] = p[0];
          A2[kt].q[1] = p[1];
        }
      }
      // layer 2: [16x256]@[256x384] + b2 -> scatter-add into aggr[target]
      const int ct = i - 16;
      v8f acc = {};
#pragma unroll
      for (int kt = 0; kt < 8; ++kt) {
        Frag B;
        const uint4* p = (const uint4*)&wbuf[cur][m * 256 + kt * 32 + hi * 16];
        B.q[0] = p[0];
        B.q[1] = p[1];
        acc = wmma_bf16(A2[kt].v, B.v, acc);
      }
      const float bias = b2[ct * 16 + m];
#pragma unroll
      for (int j = 0; j < 8; ++j) {
        const int r = j + 8 * hi;           // edge row inside tile
        const int t = __shfl(trow, r, 32);  // its target node
        atomicAdd(aggr + (long)t * 384 + ct * 16 + m, acc[j] + bias);
      }
    }
    __syncthreads();  // all waves done reading wbuf[cur]; safe to overwrite at i+2
  }
}

// ---------------- update kernel: [aggr|x] -> 3-layer MLP -> residual ----------------
// one wave = 16 nodes; block = 2 waves
__global__ __launch_bounds__(64) void k_upd(
    const float* __restrict__ aggr,          // [N,384] f32
    const unsigned short* __restrict__ xb,   // [N,128] bf16
    const float* __restrict__ x,             // [N,128] f32
    const unsigned short* __restrict__ U1T, const float* __restrict__ c1,  // [512,512]
    const unsigned short* __restrict__ U2T, const float* __restrict__ c2,  // [256,512]
    const unsigned short* __restrict__ U3T, const float* __restrict__ c3,  // [128,256]
    float* __restrict__ out, int N) {
  __shared__ unsigned short h1[2][16 * 512];
  __shared__ unsigned short h2[2][16 * 256];
  const int wave = threadIdx.x >> 5;
  const int lane = threadIdx.x & 31;
  const int m = lane & 15;
  const int hi = lane >> 4;
  const long nodeBase = ((long)blockIdx.x * 2 + wave) * 16;
  const long row = nodeBase + m;

  // A fragments: concat [aggr(384, f32->bf16) | x(128, bf16)] -> K=512
  Frag A[16];
#pragma unroll
  for (int kt = 0; kt < 16; ++kt) {
    if (kt < 12) {
      const float4* p = (const float4*)(aggr + row * 384 + kt * 32 + hi * 16);
      float4 f0 = p[0], f1 = p[1], f2 = p[2], f3 = p[3];
      A[kt].q[0] = make_uint4(packbf2(f0.x, f0.y), packbf2(f0.z, f0.w),
                              packbf2(f1.x, f1.y), packbf2(f1.z, f1.w));
      A[kt].q[1] = make_uint4(packbf2(f2.x, f2.y), packbf2(f2.z, f2.w),
                              packbf2(f3.x, f3.y), packbf2(f3.z, f3.w));
    } else {
      const uint4* p = (const uint4*)(xb + row * 128 + (kt - 12) * 32 + hi * 16);
      A[kt].q[0] = p[0];
      A[kt].q[1] = p[1];
    }
  }

  // u1: 512 -> 512, ReLU
  for (int ct = 0; ct < 32; ++ct) {
    v8f acc = {};
#pragma unroll
    for (int kt = 0; kt < 16; ++kt) {
      Frag B;
      const uint4* p = (const uint4*)(U1T + (ct * 16 + m) * 512 + kt * 32 + hi * 16);
      B.q[0] = p[0];
      B.q[1] = p[1];
      acc = wmma_bf16(A[kt].v, B.v, acc);
    }
    const float bias = c1[ct * 16 + m];
#pragma unroll
    for (int j = 0; j < 8; ++j) {
      const float v = fmaxf(acc[j] + bias, 0.f);
      h1[wave][(j + 8 * hi) * 512 + ct * 16 + m] = f2bf(v);
    }
  }
  __syncthreads();

  // u2: 512 -> 256, ReLU
  for (int ct = 0; ct < 16; ++ct) {
    v8f acc = {};
#pragma unroll
    for (int kt = 0; kt < 16; ++kt) {
      Frag Af, B;
      const uint4* pa = (const uint4*)&h1[wave][m * 512 + kt * 32 + hi * 16];
      Af.q[0] = pa[0];
      Af.q[1] = pa[1];
      const uint4* p = (const uint4*)(U2T + (ct * 16 + m) * 512 + kt * 32 + hi * 16);
      B.q[0] = p[0];
      B.q[1] = p[1];
      acc = wmma_bf16(Af.v, B.v, acc);
    }
    const float bias = c2[ct * 16 + m];
#pragma unroll
    for (int j = 0; j < 8; ++j) {
      const float v = fmaxf(acc[j] + bias, 0.f);
      h2[wave][(j + 8 * hi) * 256 + ct * 16 + m] = f2bf(v);
    }
  }
  __syncthreads();

  // u3: 256 -> 128, + residual x
  for (int ct = 0; ct < 8; ++ct) {
    v8f acc = {};
#pragma unroll
    for (int kt = 0; kt < 8; ++kt) {
      Frag Af, B;
      const uint4* pa = (const uint4*)&h2[wave][m * 256 + kt * 32 + hi * 16];
      Af.q[0] = pa[0];
      Af.q[1] = pa[1];
      const uint4* p = (const uint4*)(U3T + (ct * 16 + m) * 256 + kt * 32 + hi * 16);
      B.q[0] = p[0];
      B.q[1] = p[1];
      acc = wmma_bf16(Af.v, B.v, acc);
    }
    const float bias = c3[ct * 16 + m];
#pragma unroll
    for (int j = 0; j < 8; ++j) {
      const long nrow = nodeBase + j + 8 * hi;
      const int col = ct * 16 + m;
      out[nrow * 128 + col] = x[nrow * 128 + col] + acc[j] + bias;
    }
  }
}

extern "C" void kernel_launch(void* const* d_in, const int* in_sizes, int n_in,
                              void* d_out, int out_size, void* d_ws, size_t ws_size,
                              hipStream_t stream) {
  (void)in_sizes; (void)n_in; (void)out_size; (void)ws_size;
  const float* x  = (const float*)d_in[0];
  const int* eidx = (const int*)d_in[1];
  const float* mW1 = (const float*)d_in[2];
  const float* mb1 = (const float*)d_in[3];
  const float* mW2 = (const float*)d_in[4];
  const float* mb2 = (const float*)d_in[5];
  const float* uW1 = (const float*)d_in[6];
  const float* ub1 = (const float*)d_in[7];
  const float* uW2 = (const float*)d_in[8];
  const float* ub2 = (const float*)d_in[9];
  const float* uW3 = (const float*)d_in[10];
  const float* ub3 = (const float*)d_in[11];

  const int N = 20000, E = 640000;

  char* ws = (char*)d_ws;
  size_t off = 0;
  auto alloc = [&](size_t bytes) -> void* {
    void* p = ws + off;
    off = (off + bytes + 255) & ~(size_t)255;
    return p;
  };
  unsigned short* xb   = (unsigned short*)alloc((size_t)N * 128 * 2);
  // W1T and W2T must be contiguous: single 40-tile stream for k_msg
  unsigned short* W12T = (unsigned short*)alloc((size_t)(256 + 384) * 256 * 2);
  unsigned short* W1T  = W12T;
  unsigned short* W2T  = W12T + 256 * 256;
  unsigned short* U1T  = (unsigned short*)alloc((size_t)512 * 512 * 2);
  unsigned short* U2T  = (unsigned short*)alloc((size_t)256 * 512 * 2);
  unsigned short* U3T  = (unsigned short*)alloc((size_t)128 * 256 * 2);
  float* aggr          = (float*)alloc((size_t)N * 384 * 4);

  // prep: bf16 conversions + weight transposes + zero accumulator
  k_cvt<<<(N * 128 + 255) / 256, 256, 0, stream>>>(x, xb, N * 128);
  k_tcvt<<<(256 * 256 + 255) / 256, 256, 0, stream>>>(mW1, W1T, 256, 256);
  k_tcvt<<<(256 * 384 + 255) / 256, 256, 0, stream>>>(mW2, W2T, 256, 384);
  k_tcvt<<<(512 * 512 + 255) / 256, 256, 0, stream>>>(uW1, U1T, 512, 512);
  k_tcvt<<<(512 * 256 + 255) / 256, 256, 0, stream>>>(uW2, U2T, 512, 256);
  k_tcvt<<<(256 * 128 + 255) / 256, 256, 0, stream>>>(uW3, U3T, 256, 128);
  k_zero4<<<((N * 384 / 4) + 255) / 256, 256, 0, stream>>>((float4*)aggr, N * 384 / 4);

  // fused edge message MLP + scatter-add (wmma bf16, async LDS weight staging)
  k_msg<<<E / 16 / 4, 128, 0, stream>>>(xb, eidx, W12T, mb1, mb2, aggr, E);
  // fused node update MLP + residual (wmma bf16)
  k_upd<<<N / 16 / 2, 64, 0, stream>>>(aggr, xb, x, U1T, ub1, U2T, ub2, U3T, ub3,
                                       (float*)d_out, N);
}
